// CausalSelfAttention_88725434401111
// MI455X (gfx1250) — matrix operand order repaired and checked by reading, not compiled, not run on hardware
//
#include <hip/hip_runtime.h>
#include <math.h>

// ---------------------------------------------------------------------------
// Causal self-attention for MI455X (gfx1250), wave32 + WMMA bf16.
// fp32->bf16 convert, QKV projections (WMMA; V stored transposed [B,C,T]),
// flash attention (WMMA, online softmax in log2 domain with raw v_exp_f32,
// S^T/O^T trick so softmax stats are in-lane, v_permlanex16 half-wave merges,
// unmasked fast path + single masked diagonal block), output projection
// (WMMA, f32 out). ~88 MB of d_ws.
// ---------------------------------------------------------------------------

typedef __attribute__((ext_vector_type(16))) __bf16  v16bf;
typedef __attribute__((ext_vector_type(2)))  __bf16  v2bf;
typedef __attribute__((ext_vector_type(8)))  float   v8f;
typedef __attribute__((ext_vector_type(4)))  unsigned int v4u;

union BF16Tile {
    v16bf          v;
    unsigned short u[16];
    v4u            q[2];
};

__device__ __forceinline__ unsigned short f32_to_bf16_rte_sw(float f) {
    union { float f; unsigned int u; } x;
    x.f = f;
    unsigned int u = x.u;
    u += 0x7FFFu + ((u >> 16) & 1u);   // round-to-nearest-even
    return (unsigned short)(u >> 16);
}

// Packed f32x2 -> bf16x2 (one v_cvt_pk_bf16_f32 on CDNA5).
__device__ __forceinline__ unsigned pk_bf16(float lo, float hi) {
#if __has_builtin(__builtin_amdgcn_cvt_pk_bf16_f32)
    union { v2bf v; unsigned u; } r;
    r.v = __builtin_amdgcn_cvt_pk_bf16_f32(lo, hi);
    return r.u;
#else
    return (unsigned)f32_to_bf16_rte_sw(lo) |
           ((unsigned)f32_to_bf16_rte_sw(hi) << 16);
#endif
}

__device__ __forceinline__ unsigned short cvt1_bf16(float v) {
    return (unsigned short)pk_bf16(v, 0.f);
}

// Raw hardware exp2 (v_exp_f32): flushes tiny results to 0 -- exactly what a
// softmax wants, and 1 VALU op instead of libm's 6-op guarded sequence.
__device__ __forceinline__ float fast_exp2(float x) {
#if __has_builtin(__builtin_amdgcn_exp2f)
    return __builtin_amdgcn_exp2f(x);
#else
    return exp2f(x);
#endif
}

// Cross-half swap (lane i <-> lane i^16) as a pure VALU op (v_permlanex16_b32)
// instead of ds_bpermute + s_wait_dscnt in the softmax critical path.
__device__ __forceinline__ float xhalf_swap(float x) {
#if __has_builtin(__builtin_amdgcn_permlanex16)
    union { float f; unsigned u; } a, r;
    a.f = x;
    r.u = (unsigned)__builtin_amdgcn_permlanex16(
        (int)a.u, (int)a.u, 0x76543210, 0xfedcba98,
        /*fi=*/false, /*bound_ctrl=*/false);
    return r.f;
#else
    return __shfl_xor(x, 16, 32);
#endif
}

__device__ __forceinline__ v8f wmma_bf16(const BF16Tile& a, const BF16Tile& b, v8f c) {
    return __builtin_amdgcn_wmma_f32_16x16x32_bf16(
        /*neg_a=*/false, a.v, /*neg_b=*/false, b.v,
        /*c_mod=*/(short)0, c, /*reuse_a=*/false, /*reuse_b=*/false);
}

// Multiply a packed pair-of-bf16 word vector by a scalar (re-rounded to bf16).
__device__ __forceinline__ v4u scale_bf16x8(v4u w, float s) {
    v4u r;
#pragma unroll
    for (int i = 0; i < 4; ++i) {
        const float lo = __uint_as_float(w[i] << 16) * s;
        const float hi = __uint_as_float(w[i] & 0xffff0000u) * s;
        r[i] = pk_bf16(lo, hi);
    }
    return r;
}

// ---------------------------------------------------------------------------
// Stage 0: fp32 -> bf16 conversion (2 elements/thread, packed dword store)
// ---------------------------------------------------------------------------
__global__ void cvt_f32_to_bf16(const float* __restrict__ src,
                                unsigned int* __restrict__ dst, int npairs) {
    int i = blockIdx.x * blockDim.x + threadIdx.x;
    if (i < npairs) dst[i] = pk_bf16(src[2 * i], src[2 * i + 1]);
}

// ---------------------------------------------------------------------------
// Stage 1/3: Y[M,N] = A[M,K] @ W[N,K]^T  (nn.Linear, bias-free)
// One wave computes a 16(M) x 64(N) tile; K-loop step 32 (WMMA bf16 depth).
// OUT_MODE: 0 = f32 [M,N], 1 = bf16 [M,N],
//           2 = bf16 batch-transposed [B][N][Tdim] (vectorized b128 stores)
// ---------------------------------------------------------------------------
template <int OUT_MODE>
__global__ void gemm_abt_wmma(const unsigned short* __restrict__ A,   // [M,K] bf16
                              const unsigned short* __restrict__ W,   // [N,K] bf16
                              void* __restrict__ out,
                              int M, int N, int K, int Tdim) {
    const int lane = threadIdx.x & 31;
    const int wid  = threadIdx.x >> 5;
    const int tiles_n = N >> 6;                       // 64-wide N tiles
    const int tile = blockIdx.x * (blockDim.x >> 5) + wid;
    const int m0 = (tile / tiles_n) << 4;
    const int n0 = (tile % tiles_n) << 6;
    if (m0 >= M) return;

    const int half = lane >> 4;                       // which 16-lane half
    const int l16  = lane & 15;
    const int a_c0 = half * 8;                        // A K-chunk base
    const int b_c0 = half * 8;                        // B K-chunk base

    const v8f vzero = {0.f,0.f,0.f,0.f,0.f,0.f,0.f,0.f};
    v8f acc[4] = {vzero, vzero, vzero, vzero};

    const unsigned short* arow = A + (size_t)(m0 + l16) * K;
    for (int k = 0; k < K; k += 32) {
        BF16Tile a;
        a.q[0] = *(const v4u*)(arow + k + a_c0);          // K = a_c0 .. +7
        a.q[1] = *(const v4u*)(arow + k + a_c0 + 16);     // K = a_c0+16 .. +23
#pragma unroll
        for (int nt = 0; nt < 4; ++nt) {
            const unsigned short* wrow =
                W + (size_t)(n0 + nt * 16 + l16) * K + k + b_c0;
            BF16Tile b;
            b.q[0] = *(const v4u*)(wrow);
            b.q[1] = *(const v4u*)(wrow + 16);
            acc[nt] = wmma_bf16(a, b, acc[nt]);
        }
    }

    if (OUT_MODE == 2) {
        // bf16, per-batch transposed: out[b*N*Tdim + n*Tdim + t].
        // Lane's 8 rows are consecutive t -> one contiguous 16B store per nt.
        const int b  = m0 / Tdim;
        const int t0 = (m0 - b * Tdim) + half * 8;
        unsigned short* ob = (unsigned short*)out + (size_t)b * N * Tdim + t0;
#pragma unroll
        for (int nt = 0; nt < 4; ++nt) {
            const int n = n0 + nt * 16 + l16;
            v4u pk;
#pragma unroll
            for (int i = 0; i < 4; ++i)
                pk[i] = pk_bf16(acc[nt][2 * i], acc[nt][2 * i + 1]);
            *(v4u*)(ob + (size_t)n * Tdim) = pk;
        }
    } else {
#pragma unroll
        for (int nt = 0; nt < 4; ++nt) {
#pragma unroll
            for (int r = 0; r < 8; ++r) {
                const int m = m0 + r + half * 8;
                const int n = n0 + nt * 16 + l16;
                const float val = acc[nt][r];
                if (OUT_MODE == 1)
                    ((unsigned short*)out)[(size_t)m * N + n] = cvt1_bf16(val);
                else
                    ((float*)out)[(size_t)m * N + n] = val;
            }
        }
    }
}

// ---------------------------------------------------------------------------
// Stage 2: causal flash attention.
// ---------------------------------------------------------------------------
#define ATT_B 4
#define ATT_T 2048
#define ATT_C 1024
#define ATT_H 16
#define ATT_HD 64

// One 32-key block. Logits arrive pre-scaled (scale*log2e folded into Q), so
// softmax runs in the log2 domain on raw v_exp_f32.
// MASKED=true only for the single diagonal block per wave.
template <bool MASKED>
__device__ __forceinline__ void attn_block(
    const unsigned short* __restrict__ krow0,   // K row, tile s=0 (incl. half*8)
    const unsigned short* __restrict__ krow1,   // K row, tile s=1
    const unsigned short* __restrict__ vp0,     // V^T rows (incl. kb + half*8)
    const unsigned short* __restrict__ vp1,
    const unsigned short* __restrict__ vp2,
    const unsigned short* __restrict__ vp3,
    int kb, int qglob, int half, int l16,
    const BF16Tile& bq0, const BF16Tile& bq1,
    unsigned short (*pl)[40],                   // this wave's LDS [16][40]
    v8f* accO, float& m_run, float& l_run) {

    const v8f vzero = {0.f,0.f,0.f,0.f,0.f,0.f,0.f,0.f};

    // --- S^T = K · Qs^T : two 16-key tiles, 2 k-steps over hd=64 ---
    v8f st[2];
    {
        BF16Tile ak0, ak1;
        ak0.q[0] = *(const v4u*)(krow0);
        ak0.q[1] = *(const v4u*)(krow0 + 16);
        ak1.q[0] = *(const v4u*)(krow0 + 32);
        ak1.q[1] = *(const v4u*)(krow0 + 48);
        v8f c = vzero;
        c = wmma_bf16(ak0, bq0, c);
        st[0] = wmma_bf16(ak1, bq1, c);
    }
    {
        BF16Tile ak0, ak1;
        ak0.q[0] = *(const v4u*)(krow1);
        ak0.q[1] = *(const v4u*)(krow1 + 16);
        ak1.q[0] = *(const v4u*)(krow1 + 32);
        ak1.q[1] = *(const v4u*)(krow1 + 48);
        v8f c = vzero;
        c = wmma_bf16(ak0, bq0, c);
        st[1] = wmma_bf16(ak1, bq1, c);
    }

    if (MASKED) {
#pragma unroll
        for (int s = 0; s < 2; ++s)
#pragma unroll
            for (int r = 0; r < 8; ++r) {
                const int keyg = kb + s * 16 + r + half * 8;
                st[s][r] = (keyg <= qglob) ? st[s][r] : -INFINITY;
            }
    }

    // --- online softmax stats (per query = per lane), log2 domain ---
    float mx = -INFINITY;
#pragma unroll
    for (int s = 0; s < 2; ++s)
#pragma unroll
        for (int r = 0; r < 8; ++r) mx = fmaxf(mx, st[s][r]);
    mx = fmaxf(mx, xhalf_swap(mx));                   // merge key halves (VALU)
    const float mnew  = fmaxf(m_run, mx);
    const float alpha = fast_exp2(m_run - mnew);

    float psum = 0.f;
#pragma unroll
    for (int s = 0; s < 2; ++s)
#pragma unroll
        for (int r = 0; r < 8; ++r) {
            const float p = fast_exp2(st[s][r] - mnew);
            st[s][r] = p;
            psum += p;
        }
    psum += xhalf_swap(psum);                         // merge key halves (VALU)
    l_run = l_run * alpha + psum;
    m_run = mnew;

    // --- rescale O^T (in-lane: factor depends on q = lane) ---
#pragma unroll
    for (int dt = 0; dt < 4; ++dt)
#pragma unroll
        for (int r = 0; r < 8; ++r) accO[dt][r] *= alpha;

    // --- transpose P through LDS: packed dword stores [q][key] ---
#pragma unroll
    for (int s = 0; s < 2; ++s)
#pragma unroll
        for (int r = 0; r < 8; r += 2) {
            const int base = s * 16 + half * 8 + r;   // 4B-aligned (base even)
            *(unsigned*)&pl[l16][base] = pk_bf16(st[s][r], st[s][r + 1]);
        }

    asm volatile("s_wait_dscnt 0" ::: "memory");      // same-wave DS ordering

    BF16Tile bp;                                      // reload in B-format
    bp.q[0] = *(const v4u*)(&pl[l16][half * 8]);
    bp.q[1] = *(const v4u*)(&pl[l16][half * 8 + 16]);

    // --- O^T += V^T · P  (contiguous b128 loads from [B,C,T] layout) ---
    {
        BF16Tile av;
        av.q[0] = *(const v4u*)(vp0);
        av.q[1] = *(const v4u*)(vp0 + 16);
        accO[0] = wmma_bf16(av, bp, accO[0]);
        av.q[0] = *(const v4u*)(vp1);
        av.q[1] = *(const v4u*)(vp1 + 16);
        accO[1] = wmma_bf16(av, bp, accO[1]);
        av.q[0] = *(const v4u*)(vp2);
        av.q[1] = *(const v4u*)(vp2 + 16);
        accO[2] = wmma_bf16(av, bp, accO[2]);
        av.q[0] = *(const v4u*)(vp3);
        av.q[1] = *(const v4u*)(vp3 + 16);
        accO[3] = wmma_bf16(av, bp, accO[3]);
    }
}

__global__ void attn_wmma(const unsigned short* __restrict__ Q,   // [B,T,C]  bf16
                          const unsigned short* __restrict__ Km,  // [B,T,C]  bf16
                          const unsigned short* __restrict__ Vt,  // [B,C,T]  bf16
                          unsigned short* __restrict__ O) {       // [B,T,C]  bf16
    __shared__ __align__(16) unsigned short plds[4][16][40];      // [wave][q][key]

    const int lane  = threadIdx.x & 31;
    const int wid   = threadIdx.x >> 5;
    const int gw    = blockIdx.x * 4 + wid;
    const int qtile = gw & (ATT_T / 16 - 1);          // 128 q-tiles
    const int bh    = gw >> 7;
    const int b     = bh >> 4;
    const int h     = bh & 15;
    const int q0    = qtile << 4;

    const int half = lane >> 4;
    const int l16  = lane & 15;
    const size_t headoff = (size_t)h * ATT_HD;

    // Q^T tiles in B-format, pre-scaled by (1/sqrt(hd)) * log2(e) so the
    // softmax runs on raw v_exp_f32 with no per-element multiplies.
    const float qscale = 0.125f * 1.44269504088896340736f;
    const unsigned short* qrow =
        Q + (size_t)(b * ATT_T + q0 + l16) * ATT_C + headoff + half * 8;
    BF16Tile bq0, bq1;
    bq0.q[0] = scale_bf16x8(*(const v4u*)(qrow),      qscale);
    bq0.q[1] = scale_bf16x8(*(const v4u*)(qrow + 16), qscale);
    bq1.q[0] = scale_bf16x8(*(const v4u*)(qrow + 32), qscale);
    bq1.q[1] = scale_bf16x8(*(const v4u*)(qrow + 48), qscale);

    // Incrementing K / V^T pointers (no per-iteration 64-bit address math).
    const unsigned short* krow0 =
        Km + (size_t)(b * ATT_T + l16) * ATT_C + headoff + half * 8;
    const unsigned short* krow1 = krow0 + (size_t)16 * ATT_C;

    const size_t vtbase = (size_t)b * ATT_C * ATT_T;
    const unsigned short* vp0 =
        Vt + vtbase + (headoff + l16) * (size_t)ATT_T + half * 8;
    const unsigned short* vp1 = vp0 + 16 * (size_t)ATT_T;
    const unsigned short* vp2 = vp0 + 32 * (size_t)ATT_T;
    const unsigned short* vp3 = vp0 + 48 * (size_t)ATT_T;

    const v8f vzero = {0.f,0.f,0.f,0.f,0.f,0.f,0.f,0.f};
    v8f accO[4] = {vzero, vzero, vzero, vzero};       // O^T: lane=q, regs=d
    float m_run = -INFINITY;
    float l_run = 0.f;

    const int qglob = q0 + l16;
    const int nblocks = (q0 + 16 + 31) >> 5;          // key blocks in causal region
    const int nfull   = nblocks - 1;                  // all but the last are unmasked

    unsigned short (*pl)[40] = plds[wid];

    for (int kb0 = 0; kb0 < nfull; ++kb0) {
        attn_block<false>(krow0, krow1, vp0, vp1, vp2, vp3,
                          /*kb=*/0, qglob, half, l16, bq0, bq1, pl,
                          accO, m_run, l_run);
        krow0 += 32 * (size_t)ATT_C;
        krow1 += 32 * (size_t)ATT_C;
        vp0 += 32; vp1 += 32; vp2 += 32; vp3 += 32;
    }
    // Diagonal block (the only one that can intersect the causal boundary).
    attn_block<true>(krow0, krow1, vp0, vp1, vp2, vp3,
                     /*kb=*/nfull * 32, qglob, half, l16, bq0, bq1, pl,
                     accO, m_run, l_run);

    // --- finalize: O[q,d] = O^T / l ---
    const float linv = 1.0f / l_run;
#pragma unroll
    for (int dt = 0; dt < 4; ++dt)
#pragma unroll
        for (int r = 0; r < 8; ++r) {
            const int d = dt * 16 + r + half * 8;
            O[(size_t)(b * ATT_T + q0 + l16) * ATT_C + headoff + d] =
                cvt1_bf16(accO[dt][r] * linv);
        }
}

// ---------------------------------------------------------------------------
// Host launcher
// ---------------------------------------------------------------------------
extern "C" void kernel_launch(void* const* d_in, const int* in_sizes, int n_in,
                              void* d_out, int out_size, void* d_ws, size_t ws_size,
                              hipStream_t stream) {
    const float* x  = (const float*)d_in[0];
    const float* Wq = (const float*)d_in[1];
    const float* Wk = (const float*)d_in[2];
    const float* Wv = (const float*)d_in[3];
    const float* Wp = (const float*)d_in[4];
    float* out = (float*)d_out;

    const int B = 4, T = 2048, C = 1024;
    const int BT = B * T;                      // 8192
    const size_t nX = (size_t)BT * C;          // 8,388,608
    const size_t nW = (size_t)C * C;           // 1,048,576

    // Workspace layout (bf16 = ushort), ~88 MB total.
    unsigned short* xb  = (unsigned short*)d_ws;
    unsigned short* wqb = xb  + nX;
    unsigned short* wkb = wqb + nW;
    unsigned short* wvb = wkb + nW;
    unsigned short* wpb = wvb + nW;
    unsigned short* qb  = wpb + nW;
    unsigned short* kb  = qb  + nX;
    unsigned short* vtb = kb  + nX;            // V, transposed: [B, C, T]
    unsigned short* yb  = vtb + nX;

    // Stage 0: convert fp32 inputs to bf16 (2 elems/thread, packed stores).
    {
        const int pX = (int)(nX / 2), pW = (int)(nW / 2);
        cvt_f32_to_bf16<<<(pX + 255) / 256, 256, 0, stream>>>(x,  (unsigned*)xb,  pX);
        cvt_f32_to_bf16<<<(pW + 255) / 256, 256, 0, stream>>>(Wq, (unsigned*)wqb, pW);
        cvt_f32_to_bf16<<<(pW + 255) / 256, 256, 0, stream>>>(Wk, (unsigned*)wkb, pW);
        cvt_f32_to_bf16<<<(pW + 255) / 256, 256, 0, stream>>>(Wv, (unsigned*)wvb, pW);
        cvt_f32_to_bf16<<<(pW + 255) / 256, 256, 0, stream>>>(Wp, (unsigned*)wpb, pW);
    }

    // Stage 1: QKV projections. Q,K row-major bf16; V batch-transposed bf16.
    const int tiles  = (BT / 16) * (C / 64);   // 8192 wave-tiles
    const int blocks = tiles / 8;              // 8 waves / 256-thread block
    gemm_abt_wmma<1><<<blocks, 256, 0, stream>>>(xb, wqb, qb,  BT, C, C, T);
    gemm_abt_wmma<1><<<blocks, 256, 0, stream>>>(xb, wkb, kb,  BT, C, C, T);
    gemm_abt_wmma<2><<<blocks, 256, 0, stream>>>(xb, wvb, vtb, BT, C, C, T);

    // Stage 2: causal flash attention. 8192 waves, 4 waves / 128-thread block.
    attn_wmma<<<(4 * 16 * (T / 16)) / 4, 128, 0, stream>>>(qb, kb, vtb, yb);

    // Stage 3: output projection, f32 straight to d_out.
    gemm_abt_wmma<0><<<blocks, 256, 0, stream>>>(yb, wpb, out, BT, C, C, T);
}
